// BiLSTM_CRF_3453153706761
// MI455X (gfx1250) — matrix operand order repaired and checked by reading, not compiled, not run on hardware
//
#include <hip/hip_runtime.h>
#include <cstdint>
#include <cstddef>

#define SEQ_LEN 4096
#define NFEAT   2048
#define NHID    2048
#define H       1024
#define G4      4096   // 4*H gate rows
#define TAGS    10
#define START_TAG 8
#define STOP_TAG  9

typedef __attribute__((ext_vector_type(16))) __bf16 v16bf;
typedef __attribute__((ext_vector_type(8)))  float  v8f;

union Frag { v16bf v; uint4 q[2]; };

__device__ __forceinline__ float bf2f(unsigned short u) {
  union { unsigned u; float f; } c; c.u = ((unsigned)u) << 16; return c.f;
}
__device__ __forceinline__ unsigned short f2bf(float f) {
  union { float f; unsigned u; } c; c.f = f;
  unsigned r = (c.u + 0x7FFFu + ((c.u >> 16) & 1u)) >> 16;  // RNE
  return (unsigned short)r;
}

// ---------------------------------------------------------------- convert
__global__ void k_f32_to_bf16(const float* __restrict__ in,
                              unsigned short* __restrict__ out, int n) {
  int i = blockIdx.x * blockDim.x + threadIdx.x;
  if (i < n) out[i] = f2bf(in[i]);
}

// ---------------------------------------------------------------- init state
__global__ void k_init(float* __restrict__ hping, unsigned* __restrict__ bar) {
  int i = blockIdx.x * blockDim.x + threadIdx.x;
  if (i < 2 * 2 * H) hping[i] = 0.0f;
  if (i < 2) bar[i] = 0u;
}

// ---------------------------------------------------------------- GEMM
// out[t,g] = sum_f X[t,f]*W[g,f] + bias[g]   (NT GEMM, bf16 WMMA, f32 acc)
// 128x128 C tile / block, 256 thr = 8 waves, wave tile 32(M)x64(N).
// K pipelined in 32-wide stages: GLOBAL_LOAD_ASYNC_TO_LDS_B128 into
// double-buffered LDS, overlapped with WMMA on the other buffer;
// ASYNCcnt-throttled with s_wait_asynccnt.
__global__ __launch_bounds__(256) void k_gemm_xp(
    const unsigned short* __restrict__ X,    // [4096,2048] bf16
    const unsigned short* __restrict__ W,    // [4096,2048] bf16
    const float* __restrict__ bias,          // [4096]
    float* __restrict__ out)                 // [4096,4096] f32
{
  __shared__ unsigned short Abuf[2][128][40];  // +8 pad => 80B row stride
  __shared__ unsigned short Bbuf[2][128][40];

  const int tid  = threadIdx.x;
  const int lane = tid & 31;
  const int wave = tid >> 5;
  const int wm   = wave >> 1;      // 0..3 -> M sub-tile
  const int wn   = wave & 1;       // 0..1 -> N sub-tile
  const int half = lane >> 4;      // 0/1
  const int l16  = lane & 15;
  const int kb   = half * 8;       // A-frag K base pattern

  const int rowBase = blockIdx.y * 128;   // t
  const int colBase = blockIdx.x * 128;   // g

  const int ldr = tid >> 1;               // staging row 0..127
  const int ldc = (tid & 1) * 16;         // staging col 0 or 16

  const unsigned short* Ag = X + (size_t)(rowBase + ldr) * NFEAT + ldc;
  const unsigned short* Bg = W + (size_t)(colBase + ldr) * NFEAT + ldc;

  // LDS byte offsets (generic LDS aperture: addr[31:0] == LDS offset)
  unsigned ldsA[2], ldsB[2];
  ldsA[0] = (unsigned)(uintptr_t)&Abuf[0][ldr][ldc];
  ldsA[1] = (unsigned)(uintptr_t)&Abuf[1][ldr][ldc];
  ldsB[0] = (unsigned)(uintptr_t)&Bbuf[0][ldr][ldc];
  ldsB[1] = (unsigned)(uintptr_t)&Bbuf[1][ldr][ldc];

  // Each thread stages 32B of A and 32B of B per tile: 4 async ops/stage.
  auto stage = [&](int p, int k0) {
    const unsigned short* a = Ag + k0;
    const unsigned short* b = Bg + k0;
    asm volatile(
        "global_load_async_to_lds_b128 %0, %2, off\n\t"
        "global_load_async_to_lds_b128 %0, %2, off offset:16\n\t"
        "global_load_async_to_lds_b128 %1, %3, off\n\t"
        "global_load_async_to_lds_b128 %1, %3, off offset:16"
        :: "v"(ldsA[p]), "v"(ldsB[p]), "v"(a), "v"(b)
        : "memory");
  };

  v8f acc[2][4] = {};

  stage(0, 0);
  for (int k0 = 0; k0 < NFEAT; k0 += 32) {
    const int p = (k0 >> 5) & 1;
    if (k0 + 32 < NFEAT) {
      stage(p ^ 1, k0 + 32);                       // prefetch next tile
      asm volatile("s_wait_asynccnt 0x4" ::: "memory");  // cur tile landed
    } else {
      asm volatile("s_wait_asynccnt 0x0" ::: "memory");
    }
    __syncthreads();   // all waves' async writes for buffer p visible

    // A frags: lane holds row l16, K = {kb..kb+7} ++ {kb+16..kb+23}
    Frag af[2];
#pragma unroll
    for (int mi = 0; mi < 2; ++mi) {
      int r = wm * 32 + mi * 16 + l16;
      af[mi].q[0] = *(const uint4*)&Abuf[p][r][kb];
      af[mi].q[1] = *(const uint4*)&Abuf[p][r][kb + 16];
    }
    // B frags: lane holds col l16, K = {kb..kb+7} ++ {kb+16..kb+23}
    Frag bfr[4];
#pragma unroll
    for (int ni = 0; ni < 4; ++ni) {
      int n = wn * 64 + ni * 16 + l16;
      bfr[ni].q[0] = *(const uint4*)&Bbuf[p][n][kb];
      bfr[ni].q[1] = *(const uint4*)&Bbuf[p][n][kb + 16];
    }
#pragma unroll
    for (int ni = 0; ni < 4; ++ni)
#pragma unroll
      for (int mi = 0; mi < 2; ++mi)
        acc[mi][ni] = __builtin_amdgcn_wmma_f32_16x16x32_bf16(
            false, af[mi].v, false, bfr[ni].v, (short)0, acc[mi][ni],
            false, false);

    __syncthreads();   // buffer p fully consumed before it is re-staged
  }

  // C layout: VGPR r -> M = r + 8*half, N = l16
#pragma unroll
  for (int ni = 0; ni < 4; ++ni) {
    int col = colBase + wn * 64 + ni * 16 + l16;
    float bv = bias[col];
#pragma unroll
    for (int mi = 0; mi < 2; ++mi) {
      int rb = rowBase + wm * 32 + mi * 16 + half * 8;
#pragma unroll
      for (int r = 0; r < 8; ++r)
        out[(size_t)(rb + r) * G4 + col] = acc[mi][ni][r] + bv;
    }
  }
}

// ---------------------------------------------------------------- LSTM scan
// Persistent cooperative kernel. 128 WGs: blocks 0..63 forward, 64..127
// backward. Each WG owns 16 hidden units = 64 gate rows; its W_hh slice
// (64x1024 bf16 = 128KB) is DMA'd into LDS by the Tensor Data Mover as one
// 3D tile (4 gate groups x 16 rows x 1024 elems) and stays resident for the
// whole scan. Grid barrier per timestep via monotonic atomic counter; h
// double-buffered in global.
__global__ __launch_bounds__(256) void k_lstm_scan(
    const unsigned short* __restrict__ WhhF,
    const unsigned short* __restrict__ WhhB,
    const float* __restrict__ xpF, const float* __restrict__ xpB,
    float* __restrict__ hping,              // [dir][parity][H]
    float* __restrict__ histF, float* __restrict__ histB,
    unsigned* __restrict__ bar)             // [2]
{
  extern __shared__ unsigned char smem[];
  unsigned short* Wl = (unsigned short*)smem;            // [64][1024] 128KB
  float* hl  = (float*)(smem + 64 * 1024 * 2);           // [1024]
  float* gp  = hl + 1024;                                // [64][4] partials
  float* gs  = gp + 256;                                 // [64] gate sums
  float* cst = gs + 64;                                  // [16] cell state

  const int tid = threadIdx.x;
  const int dir = (blockIdx.x >= 64) ? 1 : 0;
  const int wg  = blockIdx.x & 63;
  const int j0  = wg * 16;

  const unsigned short* Whh = dir ? WhhB : WhhF;
  const float* xp   = dir ? xpB : xpF;
  float* hist       = dir ? histB : histF;
  float* hp         = hping + dir * 2 * H;
  unsigned* mybar   = bar + dir;

  // Weight slice preload: LDS row r <-> global gate row (r>>4)*H + j0 + (r&15)
#if __has_builtin(__builtin_amdgcn_tensor_load_to_lds)
  if (tid < 32) {   // wave 0 issues one TDM 3D tile load
    typedef __attribute__((ext_vector_type(4))) unsigned u32x4;
    typedef __attribute__((ext_vector_type(8))) int      i32x8;
    typedef __attribute__((ext_vector_type(4))) int      i32x4;
    unsigned long long ga =
        (unsigned long long)(uintptr_t)(Whh + (size_t)j0 * H);
    u32x4 g0;
    g0[0] = 1u;                                   // count=1 (valid user D#)
    g0[1] = (unsigned)(uintptr_t)Wl;              // lds_addr
    g0[2] = (unsigned)(ga & 0xFFFFFFFFu);         // global_addr[31:0]
    g0[3] = (unsigned)((ga >> 32) & 0x1FFFFFFu)   // global_addr[56:32]
          | (2u << 30);                           // type=2 ("image")
    i32x8 g1;
    g1[0] = 0x10000;                  // data_size=2B; no pad/multicast
    g1[1] = (int)(1024u << 16);       // tensor_dim0 = 1024
    g1[2] = (int)(16u << 16);         // tensor_dim1 = 16
    g1[3] = (int)(1024u << 16);       // tile_dim0 = 1024
    g1[4] = (int)(16u | (4u << 16));  // tile_dim1 = 16, tile_dim2 = 4
    g1[5] = 1024;                     // tensor_dim0_stride = H
    g1[6] = 0;                        // dim0_stride hi | dim1_stride[15:0]
    g1[7] = 0x10;                     // tensor_dim1_stride = H*H (>>16)
    i32x4 g2; g2[0] = 4; g2[1] = 0; g2[2] = 0; g2[3] = 0;  // tensor_dim2 = 4
    i32x4 g3; g3[0] = 0; g3[1] = 0; g3[2] = 0; g3[3] = 0;
    i32x8 g4;                         // extra operand on clang-23 lane
    g4[0] = 0; g4[1] = 0; g4[2] = 0; g4[3] = 0;
    g4[4] = 0; g4[5] = 0; g4[6] = 0; g4[7] = 0;
    __builtin_amdgcn_tensor_load_to_lds(g0, g1, g2, g3, g4, 0);
    asm volatile("s_wait_tensorcnt 0x0" ::: "memory");
  }
#else
  for (int idx = tid; idx < 64 * 128; idx += 256) {
    int r   = idx >> 7;
    int c8  = idx & 127;
    int grw = (r >> 4) * H + j0 + (r & 15);
    *(uint4*)&Wl[r * 1024 + c8 * 8] =
        *(const uint4*)(Whh + (size_t)grw * H + c8 * 8);
  }
#endif
  if (tid < 16) cst[tid] = 0.0f;
  __syncthreads();

  const int r    = tid >> 2;     // gate row 0..63
  const int seg  = tid & 3;      // K segment
  const int base = seg * 256;

  for (int s = 0; s < SEQ_LEN; ++s) {
    const int t   = dir ? (SEQ_LEN - 1 - s) : s;
    const int pin = s & 1;

    for (int k = tid; k < H; k += 256) hl[k] = hp[pin * H + k];
    __syncthreads();

    float acc = 0.0f;
    const uint4* wrow = (const uint4*)&Wl[r * 1024 + base];
#pragma unroll 4
    for (int kk = 0; kk < 32; ++kk) {
      uint4 q = wrow[kk];
      const float* hh = hl + base + kk * 8;
      acc += bf2f((unsigned short)(q.x & 0xffffu)) * hh[0];
      acc += bf2f((unsigned short)(q.x >> 16))     * hh[1];
      acc += bf2f((unsigned short)(q.y & 0xffffu)) * hh[2];
      acc += bf2f((unsigned short)(q.y >> 16))     * hh[3];
      acc += bf2f((unsigned short)(q.z & 0xffffu)) * hh[4];
      acc += bf2f((unsigned short)(q.z >> 16))     * hh[5];
      acc += bf2f((unsigned short)(q.w & 0xffffu)) * hh[6];
      acc += bf2f((unsigned short)(q.w >> 16))     * hh[7];
    }
    gp[r * 4 + seg] = acc;
    __syncthreads();

    if (tid < 64) {
      int grw = (tid >> 4) * H + j0 + (tid & 15);
      gs[tid] = gp[tid * 4 + 0] + gp[tid * 4 + 1] + gp[tid * 4 + 2] +
                gp[tid * 4 + 3] + xp[(size_t)t * G4 + grw];
      // prefetch next timestep's xp row segment (global_prefetch_b8)
      int tn = dir ? (t - 1) : (t + 1);
      if (tn >= 0 && tn < SEQ_LEN)
        __builtin_prefetch(&xp[(size_t)tn * G4 + grw], 0, 1);
    }
    __syncthreads();

    if (tid < 16) {
      float gi = gs[tid], gf = gs[16 + tid], gg = gs[32 + tid], go = gs[48 + tid];
      float si = 1.0f / (1.0f + __expf(-gi));
      float sf = 1.0f / (1.0f + __expf(-gf));
      float so = 1.0f / (1.0f + __expf(-go));
      float c  = sf * cst[tid] + si * tanhf(gg);
      cst[tid] = c;
      float h  = so * tanhf(c);
      hp[(pin ^ 1) * H + j0 + tid]  = h;
      hist[(size_t)t * H + j0 + tid] = h;
    }
    __syncthreads();

    if (tid == 0) {
      __threadfence();
      atomicAdd(mybar, 1u);
      unsigned target = (unsigned)(s + 1) * 64u;
      while (__hip_atomic_load(mybar, __ATOMIC_ACQUIRE,
                               __HIP_MEMORY_SCOPE_AGENT) < target) { }
    }
    __syncthreads();
  }
}

// ---------------------------------------------------------------- tag linear
__global__ __launch_bounds__(320) void k_linear(
    const float* __restrict__ histF, const float* __restrict__ histB,
    const float* __restrict__ Wlin, const float* __restrict__ blin,
    float* __restrict__ feats)
{
  const int t    = blockIdx.x;
  const int wave = threadIdx.x >> 5;   // tag
  const int lane = threadIdx.x & 31;
  const float* w = Wlin + wave * NHID;
  float acc = 0.0f;
  for (int k = lane; k < NHID; k += 32) {
    float x = (k < H) ? histF[(size_t)t * H + k] : histB[(size_t)t * H + (k - H)];
    acc += x * w[k];
  }
#pragma unroll
  for (int off = 16; off; off >>= 1) acc += __shfl_xor(acc, off, 32);
  if (lane == 0) feats[t * TAGS + wave] = acc + blin[wave];
}

// ---------------------------------------------------------------- viterbi
__global__ void k_viterbi(const float* __restrict__ feats,
                          const float* __restrict__ trans,
                          int* __restrict__ bps,
                          float* __restrict__ out)
{
  __shared__ float fv[TAGS];
  __shared__ float nv[TAGS];
  __shared__ float term[TAGS];
  const int j = threadIdx.x;
  float tr[TAGS];
  if (j < TAGS) {
    for (int p = 0; p < TAGS; ++p) tr[p] = trans[j * TAGS + p];
    fv[j] = (j == START_TAG) ? 0.0f : -10000.0f;
  }
  __syncthreads();
  for (int t = 0; t < SEQ_LEN; ++t) {
    if (j < TAGS) {
      float m = -3.4e38f; int arg = 0;
      for (int p = 0; p < TAGS; ++p) {
        float sc = fv[p] + tr[p];
        if (sc > m) { m = sc; arg = p; }
      }
      nv[j] = m + feats[t * TAGS + j];
      bps[t * TAGS + j] = arg;
    }
    __syncthreads();
    if (j < TAGS) fv[j] = nv[j];
    __syncthreads();
  }
  if (j < TAGS) term[j] = fv[j] + trans[STOP_TAG * TAGS + j];
  __syncthreads();
  if (j == 0) {
    float m = term[0]; int best = 0;
    for (int p = 1; p < TAGS; ++p) if (term[p] > m) { m = term[p]; best = p; }
    out[0] = m;
    int tag = best;
    out[1 + SEQ_LEN - 1] = (float)tag;
    for (int t = SEQ_LEN - 1; t >= 1; --t) {
      tag = bps[t * TAGS + tag];
      out[1 + t - 1] = (float)tag;
    }
  }
}

// ---------------------------------------------------------------- launcher
extern "C" void kernel_launch(void* const* d_in, const int* in_sizes, int n_in,
                              void* d_out, int out_size, void* d_ws, size_t ws_size,
                              hipStream_t stream)
{
  (void)in_sizes; (void)n_in; (void)out_size; (void)ws_size;
  const float* sentence = (const float*)d_in[0];
  const float* W_ih_f   = (const float*)d_in[1];
  const float* W_hh_f   = (const float*)d_in[2];
  const float* b_f      = (const float*)d_in[3];
  const float* W_ih_b   = (const float*)d_in[4];
  const float* W_hh_b   = (const float*)d_in[5];
  const float* b_b      = (const float*)d_in[6];
  const float* W_lin    = (const float*)d_in[7];
  const float* b_lin    = (const float*)d_in[8];
  const float* trans    = (const float*)d_in[9];

  unsigned char* ws = (unsigned char*)d_ws;
  size_t off = 0;
  auto suballoc = [&](size_t bytes) -> unsigned char* {
    unsigned char* p = ws + off;
    off += (bytes + 255) & ~(size_t)255;
    return p;
  };
  unsigned short* Xbf   = (unsigned short*)suballoc((size_t)SEQ_LEN * NFEAT * 2);
  unsigned short* Wfbf  = (unsigned short*)suballoc((size_t)G4 * NFEAT * 2);
  unsigned short* Wbbf  = (unsigned short*)suballoc((size_t)G4 * NFEAT * 2);
  unsigned short* WhFbf = (unsigned short*)suballoc((size_t)G4 * H * 2);
  unsigned short* WhBbf = (unsigned short*)suballoc((size_t)G4 * H * 2);
  float* xpF   = (float*)suballoc((size_t)SEQ_LEN * G4 * 4);
  float* xpB   = (float*)suballoc((size_t)SEQ_LEN * G4 * 4);
  float* histF = (float*)suballoc((size_t)SEQ_LEN * H * 4);
  float* histB = (float*)suballoc((size_t)SEQ_LEN * H * 4);
  float* hping = (float*)suballoc(2 * 2 * H * 4);
  float* feats = (float*)suballoc((size_t)SEQ_LEN * TAGS * 4);
  int*   bps   = (int*)suballoc((size_t)SEQ_LEN * TAGS * 4);
  unsigned* bar = (unsigned*)suballoc(256);

  int n;
  n = SEQ_LEN * NFEAT; k_f32_to_bf16<<<(n + 255) / 256, 256, 0, stream>>>(sentence, Xbf, n);
  n = G4 * NFEAT;      k_f32_to_bf16<<<(n + 255) / 256, 256, 0, stream>>>(W_ih_f, Wfbf, n);
  n = G4 * NFEAT;      k_f32_to_bf16<<<(n + 255) / 256, 256, 0, stream>>>(W_ih_b, Wbbf, n);
  n = G4 * H;          k_f32_to_bf16<<<(n + 255) / 256, 256, 0, stream>>>(W_hh_f, WhFbf, n);
  n = G4 * H;          k_f32_to_bf16<<<(n + 255) / 256, 256, 0, stream>>>(W_hh_b, WhBbf, n);
  k_init<<<16, 256, 0, stream>>>(hping, bar);

  dim3 gg(G4 / 128, SEQ_LEN / 128);
  k_gemm_xp<<<gg, 256, 0, stream>>>(Xbf, Wfbf, b_f, xpF);
  k_gemm_xp<<<gg, 256, 0, stream>>>(Xbf, Wbbf, b_b, xpB);

  size_t smem = 64 * 1024 * 2 + 1024 * 4 + 256 * 4 + 64 * 4 + 16 * 4;
  k_lstm_scan<<<128, 256, smem, stream>>>(WhFbf, WhBbf, xpF, xpB,
                                          hping, histF, histB, bar);

  k_linear<<<SEQ_LEN, 320, 0, stream>>>(histF, histB, W_lin, b_lin, feats);
  k_viterbi<<<1, 32, 0, stream>>>(feats, trans, bps, (float*)d_out);
}
